// MambaBlock_40209483825877
// MI455X (gfx1250) — compile-verified, matrix operand order checked by reading, not run on hardware
//
#include <hip/hip_runtime.h>

// ---------------- problem constants ----------------
#define BATCH   8
#define SEQ     4096
#define DM      768
#define DIN     1536          // 2*DM
#define DXZ     3072          // 2*DIN
#define DST     64
#define DTR     48
#define DBLW    176           // DTR + 2*DST
#define NTOK    (BATCH*SEQ)   // 32768
#define LN_EPS  1e-5f

typedef __attribute__((ext_vector_type(16))) __bf16 v16bf;
typedef __attribute__((ext_vector_type(8)))  __bf16 v8bf;
typedef __attribute__((ext_vector_type(8)))  float  v8f;

// ---------------- CDNA5 async global->LDS copy (16B per lane) ----------------
// GLOBAL_LOAD_ASYNC_TO_LDS_B128: LDS[vdst_lane] = MEM[vaddr_lane]; ASYNCcnt-tracked.
__device__ __forceinline__ void async_cp_b128(void* lds_ptr, const void* gptr) {
    uint32_t lds = (uint32_t)(size_t)lds_ptr;     // flat LDS addr: low 32 bits = LDS byte offset
    uint64_t g   = (uint64_t)(size_t)gptr;
    asm volatile("global_load_async_to_lds_b128 %0, %1, off"
                 :: "v"(lds), "v"(g) : "memory");
}
__device__ __forceinline__ void wait_async3() {
    asm volatile("s_wait_asynccnt 0x3" ::: "memory");
}
__device__ __forceinline__ void wait_async0() {
    asm volatile("s_wait_asynccnt 0x0" ::: "memory");
}

// ---------------- fp32 -> bf16 cast ----------------
__global__ void k_cast_bf16(const float* __restrict__ in, __bf16* __restrict__ out, int n) {
    int i = blockIdx.x * 256 + threadIdx.x;
    if (i < n) out[i] = (__bf16)in[i];
}

// ---------------- LayerNorm (one token per block) ----------------
__global__ __launch_bounds__(256) void k_layernorm(const float* __restrict__ x,
                                                   const float* __restrict__ w,
                                                   const float* __restrict__ b,
                                                   __bf16* __restrict__ out) {
    int t = blockIdx.x;
    const float* row = x + (size_t)t * DM;
    float v[3], s = 0.f, s2 = 0.f;
#pragma unroll
    for (int k = 0; k < 3; ++k) {
        v[k] = row[threadIdx.x + k * 256];
        s += v[k]; s2 += v[k] * v[k];
    }
#pragma unroll
    for (int m = 16; m >= 1; m >>= 1) {
        s  += __shfl_xor(s,  m, 32);
        s2 += __shfl_xor(s2, m, 32);
    }
    __shared__ float ps[8], ps2[8];
    int wid = threadIdx.x >> 5, lane = threadIdx.x & 31;
    if (lane == 0) { ps[wid] = s; ps2[wid] = s2; }
    __syncthreads();
    s = 0.f; s2 = 0.f;
#pragma unroll
    for (int i = 0; i < 8; ++i) { s += ps[i]; s2 += ps2[i]; }
    float mu  = s * (1.f / DM);
    float var = s2 * (1.f / DM) - mu * mu;
    float rs  = rsqrtf(var + LN_EPS);
    __bf16* orow = out + (size_t)t * DM;
#pragma unroll
    for (int k = 0; k < 3; ++k) {
        int c = threadIdx.x + k * 256;
        orow[c] = (__bf16)((v[k] - mu) * rs * w[c] + b[c]);
    }
}

// ---------------- bf16 WMMA GEMM: C[M,N] = A[M,K] * W[N,K]^T (+res) ----------------
// block tile 128(M) x 64(N), K-step 32 (== one wmma_f32_16x16x32_bf16)
// 8 waves: 4 along M x 2 along N, each wave -> 32x32 (four 16x16 accumulators)
// Double-buffered LDS filled by global_load_async_to_lds_b128 (3 per thread/tile).
// Requires M % 128 == 0 (true for all launches); ragged N handled by clamping
// B-row reads (duplicates computed, stores guarded).
#define LDK 40   // padded LDS row stride (elems): 80B, 16B-aligned, conflict-free

__device__ __forceinline__ v16bf ld_frag(const __bf16* smem, int row0, int lane) {
    // ISA 7.12.2 bf16 A/B layout: lane r=lane&15 holds row (M or N) r;
    // contiguous 8 elems at K = kb, 8 more at K = kb+16, kb = (lane>>4)*8.
    int r  = row0 + (lane & 15);
    int kb = (lane >> 4) << 3;
    const __bf16* p = smem + r * LDK + kb;
    v8bf lo = *(const v8bf*)(p);
    v8bf hi = *(const v8bf*)(p + 16);
    v16bf f;
#pragma unroll
    for (int i = 0; i < 8; ++i) { f[i] = lo[i]; f[i + 8] = hi[i]; }
    return f;
}

__device__ __forceinline__ void stage_tile(const __bf16* __restrict__ A,
                                           const __bf16* __restrict__ W,
                                           __bf16* Asb, __bf16* Bsb,
                                           int tid, int m0, int n0, int nClamp,
                                           int K, int k0) {
    // A tile: 128 rows x 32 cols = 512 v8bf chunks -> 2 per thread
    {
        int r0 = tid >> 2, c0 = (tid & 3) << 3;
        async_cp_b128(Asb + r0 * LDK + c0, A + (size_t)(m0 + r0) * K + k0 + c0);
        int i = tid + 256;
        int r1 = i >> 2, c1 = (i & 3) << 3;
        async_cp_b128(Asb + r1 * LDK + c1, A + (size_t)(m0 + r1) * K + k0 + c1);
    }
    // B tile: 64 rows x 32 cols = 256 chunks -> 1 per thread (row clamped for ragged N)
    {
        int r = tid >> 2, c = (tid & 3) << 3;
        int gr = n0 + r;
        if (gr > nClamp) gr = nClamp;
        async_cp_b128(Bsb + r * LDK + c, W + (size_t)gr * K + k0 + c);
    }
}

__global__ __launch_bounds__(256) void k_gemm_bf16(const __bf16* __restrict__ A,
                                                   const __bf16* __restrict__ W,
                                                   float* __restrict__ C,
                                                   const float* __restrict__ res,
                                                   int M, int N, int K) {
    __shared__ __bf16 As[2][128 * LDK];
    __shared__ __bf16 Bs[2][64 * LDK];
    int tid  = threadIdx.x;
    int lane = tid & 31;
    int wave = tid >> 5;
    int wm = wave & 3;       // 0..3 -> M sub-tile
    int wn = wave >> 2;      // 0..1 -> N sub-tile
    int m0 = blockIdx.y * 128;
    int n0 = blockIdx.x * 64;
    int nClamp = N - 1;

    v8f acc[2][2] = {};

    const int KT = K >> 5;   // K-tiles of 32
    stage_tile(A, W, As[0], Bs[0], tid, m0, n0, nClamp, K, 0);

    for (int kt = 0; kt < KT; ++kt) {
        const __bf16* Asb = As[kt & 1];
        const __bf16* Bsb = Bs[kt & 1];
        if (kt + 1 < KT) {
            stage_tile(A, W, As[(kt + 1) & 1], Bs[(kt + 1) & 1],
                       tid, m0, n0, nClamp, K, (kt + 1) << 5);
            wait_async3();   // newest 3 may be in flight; current tile's 3 are done
        } else {
            wait_async0();
        }
        __syncthreads();     // current buffer visible to all waves

        v16bf a0 = ld_frag(Asb, wm * 32,      lane);
        v16bf a1 = ld_frag(Asb, wm * 32 + 16, lane);
        v16bf b0 = ld_frag(Bsb, wn * 32,      lane);
        v16bf b1 = ld_frag(Bsb, wn * 32 + 16, lane);

        acc[0][0] = __builtin_amdgcn_wmma_f32_16x16x32_bf16(false, a0, false, b0, (short)0, acc[0][0], false, false);
        acc[0][1] = __builtin_amdgcn_wmma_f32_16x16x32_bf16(false, a0, false, b1, (short)0, acc[0][1], false, false);
        acc[1][0] = __builtin_amdgcn_wmma_f32_16x16x32_bf16(false, a1, false, b0, (short)0, acc[1][0], false, false);
        acc[1][1] = __builtin_amdgcn_wmma_f32_16x16x32_bf16(false, a1, false, b1, (short)0, acc[1][1], false, false);
        __syncthreads();     // all reads of this buffer done before it is re-staged
    }

    // epilogue: C/D layout — lane n = lane&15, VGPR v holds M = (lane>>4)*8 + v
    int nn = lane & 15, mb = (lane >> 4) << 3;
#pragma unroll
    for (int tm = 0; tm < 2; ++tm)
#pragma unroll
        for (int tn = 0; tn < 2; ++tn) {
            int col   = n0 + wn * 32 + tn * 16 + nn;
            int rbase = m0 + wm * 32 + tm * 16 + mb;
#pragma unroll
            for (int v = 0; v < 8; ++v) {
                int row = rbase + v;
                if (col < N) {
                    float val = acc[tm][tn][v];
                    size_t idx = (size_t)row * N + col;
                    if (res) val += res[idx];
                    C[idx] = val;
                }
            }
        }
}

// ---------------- depthwise causal conv1d (k=4) + SiLU ----------------
__global__ void k_conv_silu(const float* __restrict__ xz,
                            const float* __restrict__ cw,
                            const float* __restrict__ cb,
                            float* __restrict__ xs,
                            __bf16* __restrict__ xsb) {
    size_t i = (size_t)blockIdx.x * 256 + threadIdx.x;
    int d = (int)(i % DIN);
    int t = (int)(i / DIN);
    int b = t >> 12;          // / SEQ
    int l = t & (SEQ - 1);
    float acc = cb[d];
#pragma unroll
    for (int j = 0; j < 4; ++j) {
        int l2 = l - 3 + j;
        if (l2 >= 0)
            acc += cw[d * 4 + j] * xz[(size_t)(b * SEQ + l2) * DXZ + d];
    }
    float s = acc / (1.f + __expf(-acc));   // SiLU
    xs[i]  = s;
    xsb[i] = (__bf16)s;
}

// ---------------- dt = softplus(dtp @ dt_proj_w^T + b) ----------------
__global__ __launch_bounds__(256) void k_dt(const float* __restrict__ dbl,
                                            const float* __restrict__ dtw,
                                            const float* __restrict__ dtb,
                                            float* __restrict__ dt) {
    int t = blockIdx.y;
    int d = blockIdx.x * 256 + threadIdx.x;
    __shared__ float sp[DTR];
    if (threadIdx.x < DTR) sp[threadIdx.x] = dbl[(size_t)t * DBLW + threadIdx.x];
    __syncthreads();
    float acc = dtb[d];
#pragma unroll
    for (int r = 0; r < DTR; ++r) acc += sp[r] * dtw[d * DTR + r];
    dt[(size_t)t * DIN + d] = (acc > 20.f) ? acc : log1pf(__expf(acc));
}

// ---------------- selective scan: one wave per (batch, channel) ----------------
// lane holds states n=lane and n=lane+32; y reduced via shfl_xor over wave32.
__global__ __launch_bounds__(256) void k_scan(const float* __restrict__ dt,
                                              const float* __restrict__ xs,
                                              const float* __restrict__ dbl,
                                              const float* __restrict__ A_log,
                                              float* __restrict__ ys) {
    int wave = threadIdx.x >> 5;
    int lane = threadIdx.x & 31;
    int d = blockIdx.x * 8 + wave;
    int b = blockIdx.y;

    float A0 = -__expf(A_log[d * DST + lane]);
    float A1 = -__expf(A_log[d * DST + 32 + lane]);
    float h0 = 0.f, h1 = 0.f;

    const float* dtp = dt  + (size_t)b * SEQ * DIN + d;
    const float* xp  = xs  + (size_t)b * SEQ * DIN + d;
    const float* Bp  = dbl + (size_t)b * SEQ * DBLW + DTR;        // B block
    const float* Cp  = Bp + DST;                                   // C block
    float* yo        = ys  + (size_t)b * SEQ * DIN + d;

    for (int l0 = 0; l0 < SEQ; l0 += 32) {
        float dtv = dtp[(size_t)(l0 + lane) * DIN];
        float xv  = xp [(size_t)(l0 + lane) * DIN];
#pragma unroll 4
        for (int i = 0; i < 32; ++i) {
            float dt_t = __shfl(dtv, i, 32);
            float x_t  = __shfl(xv,  i, 32);
            size_t trow = (size_t)(l0 + i) * DBLW;
            float B0 = Bp[trow + lane];
            float B1 = Bp[trow + 32 + lane];
            float C0 = Cp[trow + lane];
            float C1 = Cp[trow + 32 + lane];
            float dx = dt_t * x_t;
            h0 = __expf(dt_t * A0) * h0 + dx * B0;
            h1 = __expf(dt_t * A1) * h1 + dx * B1;
            float y = h0 * C0 + h1 * C1;
#pragma unroll
            for (int m = 16; m >= 1; m >>= 1) y += __shfl_xor(y, m, 32);
            if (lane == 0) yo[(size_t)(l0 + i) * DIN] = y;
        }
    }
}

// ---------------- y = (ys + xs*D) * silu(z), z = xz[:,DIN:] ----------------
__global__ void k_gate(const float* __restrict__ ys,
                       const float* __restrict__ xs,
                       const float* __restrict__ xz,
                       const float* __restrict__ Dv,
                       __bf16* __restrict__ yb) {
    size_t i = (size_t)blockIdx.x * 256 + threadIdx.x;
    int d = (int)(i % DIN);
    int t = (int)(i / DIN);
    float z  = xz[(size_t)t * DXZ + DIN + d];
    float sz = z / (1.f + __expf(-z));
    float y  = (ys[i] + xs[i] * Dv[d]) * sz;
    yb[i] = (__bf16)y;
}

// ---------------- launcher ----------------
extern "C" void kernel_launch(void* const* d_in, const int* in_sizes, int n_in,
                              void* d_out, int out_size, void* d_ws, size_t ws_size,
                              hipStream_t stream) {
    const float* x        = (const float*)d_in[0];
    const float* ln_w     = (const float*)d_in[1];
    const float* ln_b     = (const float*)d_in[2];
    const float* in_proj  = (const float*)d_in[3];   // [3072, 768]
    const float* conv_w   = (const float*)d_in[4];   // [1536, 1, 4]
    const float* conv_b   = (const float*)d_in[5];
    const float* x_proj   = (const float*)d_in[6];   // [176, 1536]
    const float* dt_projw = (const float*)d_in[7];   // [1536, 48]
    const float* dt_projb = (const float*)d_in[8];
    const float* A_log    = (const float*)d_in[9];   // [1536, 64]
    const float* Dv       = (const float*)d_in[10];
    const float* out_proj = (const float*)d_in[11];  // [768, 1536]
    float* out            = (float*)d_out;

    char* ws = (char*)d_ws;
    size_t o = 0;
    auto alloc = [&](size_t bytes) { size_t r = o; o += (bytes + 255) & ~(size_t)255; return r; };

    __bf16* w_in_bf = (__bf16*)(ws + alloc((size_t)DXZ * DM * 2));
    __bf16* w_xp_bf = (__bf16*)(ws + alloc((size_t)DBLW * DIN * 2));
    __bf16* w_op_bf = (__bf16*)(ws + alloc((size_t)DM * DIN * 2));
    __bf16* xn_bf   = (__bf16*)(ws + alloc((size_t)NTOK * DM * 2));
    float*  xz      = (float*) (ws + alloc((size_t)NTOK * DXZ * 4));
    float*  xs      = (float*) (ws + alloc((size_t)NTOK * DIN * 4));
    __bf16* xs_bf   = (__bf16*)(ws + alloc((size_t)NTOK * DIN * 2));
    float*  dbl     = (float*) (ws + alloc((size_t)NTOK * DBLW * 4));
    float*  dt      = (float*) (ws + alloc((size_t)NTOK * DIN * 4));
    float*  ysb     = (float*) (ws + alloc((size_t)NTOK * DIN * 4));
    __bf16* y_bf    = (__bf16*)(ws + alloc((size_t)NTOK * DIN * 2));

    // weight casts (fp32 -> bf16, layouts kept as [N,K])
    k_cast_bf16<<<(DXZ * DM + 255) / 256, 256, 0, stream>>>(in_proj, w_in_bf, DXZ * DM);
    k_cast_bf16<<<(DBLW * DIN + 255) / 256, 256, 0, stream>>>(x_proj, w_xp_bf, DBLW * DIN);
    k_cast_bf16<<<(DM * DIN + 255) / 256, 256, 0, stream>>>(out_proj, w_op_bf, DM * DIN);

    // 1) LayerNorm -> bf16 activations
    k_layernorm<<<NTOK, 256, 0, stream>>>(x, ln_w, ln_b, xn_bf);

    // 2) in_proj GEMM: [32768,768] x [3072,768]^T -> xz
    k_gemm_bf16<<<dim3(DXZ / 64, NTOK / 128), 256, 0, stream>>>(
        xn_bf, w_in_bf, xz, nullptr, NTOK, DXZ, DM);

    // 3) depthwise causal conv + SiLU
    k_conv_silu<<<(size_t)NTOK * DIN / 256, 256, 0, stream>>>(xz, conv_w, conv_b, xs, xs_bf);

    // 4) x_proj GEMM: [32768,1536] x [176,1536]^T -> dbl
    k_gemm_bf16<<<dim3((DBLW + 63) / 64, NTOK / 128), 256, 0, stream>>>(
        xs_bf, w_xp_bf, dbl, nullptr, NTOK, DBLW, DIN);

    // 5) dt projection + softplus
    k_dt<<<dim3(DIN / 256, NTOK), 256, 0, stream>>>(dbl, dt_projw, dt_projb, dt);

    // 6) selective scan
    k_scan<<<dim3(DIN / 8, BATCH), 256, 0, stream>>>(dt, xs, dbl, A_log, ysb);

    // 7) gate
    k_gate<<<(size_t)NTOK * DIN / 256, 256, 0, stream>>>(ysb, xs, xz, Dv, y_bf);

    // 8) out_proj GEMM + residual: out = x + y @ out_proj^T
    k_gemm_bf16<<<dim3(DM / 64, NTOK / 128), 256, 0, stream>>>(
        y_bf, w_op_bf, out, x, NTOK, DM, DIN);
}
